// CrossAttentionBlock_66597762892025
// MI455X (gfx1250) — matrix operand order repaired and checked by reading, not compiled, not run on hardware
//
#include <hip/hip_runtime.h>

// ---------------------------------------------------------------------------
// CDNA5 (gfx1250) cross-attention block: bf16 WMMA + TDM-fed GEMM tiles.
// ---------------------------------------------------------------------------

typedef __attribute__((ext_vector_type(8)))  float    v8f;
typedef __attribute__((ext_vector_type(16))) __bf16   v16bf;
typedef __attribute__((ext_vector_type(8)))  __bf16   v8bf;
typedef __attribute__((ext_vector_type(4)))  unsigned v4u;
typedef __attribute__((ext_vector_type(8)))  int      v8i;
typedef __attribute__((ext_vector_type(4)))  int      v4i;

#define Hdim 768
#define NHEAD 12
#define HDIM 64
#define FFDIM 3072
#define BATCH 8
#define SEQ 1024
#define MROWS (BATCH * SEQ)   // 8192
#define LDSA 40               // LDS row stride in bf16 elems (80B = 64B tile row + 16B pad)

__device__ __forceinline__ unsigned short f32_to_bf16(float f) {
  union { float f; unsigned u; } x; x.f = f;
  unsigned u = x.u;
  unsigned r = (u + 0x7fffu + ((u >> 16) & 1u)) >> 16;  // RNE
  return (unsigned short)r;
}

__device__ __forceinline__ v8f zero8() {
  v8f z;
#pragma unroll
  for (int i = 0; i < 8; ++i) z[i] = 0.f;
  return z;
}

__device__ __forceinline__ v16bf pack16(v8bf lo, v8bf hi) {
  v16bf r;
#pragma unroll
  for (int i = 0; i < 8; ++i) { r[i] = lo[i]; r[i + 8] = hi[i]; }
  return r;
}

__device__ __forceinline__ v8f wmma_bf16(v16bf a, v16bf b, v8f c) {
  return __builtin_amdgcn_wmma_f32_16x16x32_bf16(false, a, false, b, (short)0, c,
                                                 false, false);
}

// ---------------------------------------------------------------------------
// Tensor Data Mover: DMA a 2-D tile (rows x 32 bf16) from global into LDS.
// D# packing per CDNA5 ISA:
//   group0: count=1 | lds_addr | global_addr[56:0] | type=2
//   group1: data_size=1(2B), pad_enable, pad_interval=3 (every 16 dwords),
//           pad_amount=3 (4 dwords)  -> LDS row stride 64B+16B = LDSA
//           tensor_dim0=K, tensor_dim1=rows, tile_dim0=32, tile_dim1=rows,
//           tensor_dim0_stride=K
// clang-23 toolchain: 6-arg builtin (g0, g1, g2, g3, extra v8i, cpol).
// ---------------------------------------------------------------------------
__device__ __forceinline__ void tdm_load_tile(unsigned lds_off,
                                              const unsigned short* gptr,
                                              int rows, int K) {
  unsigned long long ga = (unsigned long long)(size_t)gptr;
  v4u g0;
  g0[0] = 1u;                                   // count=1, user mode
  g0[1] = lds_off;                              // LDS byte address
  g0[2] = (unsigned)(ga & 0xffffffffu);         // global_addr[31:0]
  g0[3] = (unsigned)((ga >> 32) & 0x01ffffffu)  // global_addr[56:32]
          | (2u << 30);                         // type=2 ("image")
  v8i g1;
  g1[0] = (int)((1u << 16)      // data_size = 2 bytes
                | (1u << 20)    // pad_enable
                | (3u << 22)    // pad_interval: 16 dwords (64B)
                | (3u << 25));  // pad_amount: 4 dwords (16B)
  g1[1] = (int)(((unsigned)K & 0xffffu) << 16);                  // tensor_dim0 lo
  g1[2] = (int)((((unsigned)K >> 16) & 0xffffu) |
                (((unsigned)rows & 0xffffu) << 16));             // td0 hi | td1 lo
  g1[3] = (int)(32u << 16);                                      // td1 hi=0 | tile_dim0=32
  g1[4] = (int)((unsigned)rows & 0xffffu);                       // tile_dim1 | tile_dim2=0
  g1[5] = (int)K;                                                // tensor_dim0_stride lo
  g1[6] = 0;                                                     // stride hi | td1stride lo
  g1[7] = 0;
  v4i z4 = {0, 0, 0, 0};
  v8i z8 = {0, 0, 0, 0, 0, 0, 0, 0};
  __builtin_amdgcn_tensor_load_to_lds(g0, g1, z4, z4, z8, 0);
}

// ---------------------------------------------------------------------------
// fp32 -> bf16 convert (grid-stride)
// ---------------------------------------------------------------------------
__global__ void cvt_bf16(const float* __restrict__ x,
                         unsigned short* __restrict__ y, int n) {
  for (int i = blockIdx.x * blockDim.x + threadIdx.x; i < n;
       i += gridDim.x * blockDim.x)
    y[i] = f32_to_bf16(x[i]);
}

// W[K][N] fp32 -> Wt[N][K] bf16 (so GEMM B-operands are contiguous along K)
__global__ void tcvt_bf16(const float* __restrict__ W,
                          unsigned short* __restrict__ Wt, int K, int N) {
  int total = K * N;
  for (int i = blockIdx.x * blockDim.x + threadIdx.x; i < total;
       i += gridDim.x * blockDim.x) {
    int k = i / N, n = i - k * N;
    Wt[(size_t)n * K + k] = f32_to_bf16(W[i]);
  }
}

// ---------------------------------------------------------------------------
// Tiled bf16 GEMM:  C[M][N] = A[M][K] @ Bt[N][K]^T + bias
// block: 256 thr = 8 waves (4 along M x 2 along N); block tile 256(M) x 128(N)
// per-wave 64x64 = 4x4 WMMA tiles; K-step 32; LDS double-buffered, fed by TDM.
// MODE 0: f32 out   MODE 2: relu->bf16   MODE 3: bf16 [bh][s][64]   MODE 4: bf16 [bh][d][s]
// ---------------------------------------------------------------------------
template <int MODE>
__global__ __launch_bounds__(256) void gemm_bf16(
    const unsigned short* __restrict__ A, const unsigned short* __restrict__ Bt,
    const float* __restrict__ bias, void* __restrict__ Cout, int M, int N, int K) {
  __shared__ __align__(16) unsigned short lA[2][256 * LDSA];  // 40.0 KB
  __shared__ __align__(16) unsigned short lB[2][128 * LDSA];  // 20.0 KB

  const int t    = threadIdx.x;
  const int lane = t & 31;
  const int w    = t >> 5;
  const int wm   = w & 3;    // 4 waves along M (64 rows each)
  const int wn   = w >> 2;   // 2 waves along N (64 cols each)
  const int l16  = lane & 15;
  const int hf   = lane >> 4;
  const int m0   = blockIdx.y * 256;
  const int n0   = blockIdx.x * 128;

  v8f acc[4][4];
#pragma unroll
  for (int i = 0; i < 4; ++i)
#pragma unroll
    for (int j = 0; j < 4; ++j) acc[i][j] = zero8();

  // prologue: DMA tile 0 into buffer 0 (wave 0 owns the TDM queue)
  if (w == 0) {
    tdm_load_tile((unsigned)(size_t)&lA[0][0], A + (size_t)m0 * K, 256, K);
    tdm_load_tile((unsigned)(size_t)&lB[0][0], Bt + (size_t)n0 * K, 128, K);
  }

  for (int k0 = 0; k0 < K; k0 += 32) {
    const int buf = (k0 >> 5) & 1;
    const bool more = (k0 + 32) < K;
    if (w == 0) {
      if (more) {  // prefetch next K-tile into the other buffer
        tdm_load_tile((unsigned)(size_t)&lA[buf ^ 1][0],
                      A + (size_t)m0 * K + (k0 + 32), 256, K);
        tdm_load_tile((unsigned)(size_t)&lB[buf ^ 1][0],
                      Bt + (size_t)n0 * K + (k0 + 32), 128, K);
        // tensor ops complete in order: <=2 outstanding => current buf ready
        __builtin_amdgcn_s_wait_tensorcnt(2);
      } else {
        __builtin_amdgcn_s_wait_tensorcnt(0);
      }
    }
    __syncthreads();

    // ---- WMMA operand gathers per ISA VGPR layouts ----
    v16bf Aop[4];
#pragma unroll
    for (int mt = 0; mt < 4; ++mt) {
      const unsigned short* p = &lA[buf][(wm * 64 + mt * 16 + l16) * LDSA];
      v8bf lo = *(const v8bf*)(p + hf * 8);        // K = hf*8 .. +7
      v8bf hi = *(const v8bf*)(p + 16 + hf * 8);   // K = 16+hf*8 .. +7
      Aop[mt] = pack16(lo, hi);
    }
#pragma unroll
    for (int nt = 0; nt < 4; ++nt) {
      const unsigned short* p = &lB[buf][(wn * 64 + nt * 16 + l16) * LDSA];
      v8bf lo = *(const v8bf*)(p + hf * 8);        // K = hf*8 .. +7
      v8bf hi = *(const v8bf*)(p + 16 + hf * 8);   // K = 16+hf*8 .. +7
      v16bf Bop = pack16(lo, hi);
#pragma unroll
      for (int mt = 0; mt < 4; ++mt)
        acc[mt][nt] = wmma_bf16(Aop[mt], Bop, acc[mt][nt]);
    }
    __syncthreads();  // reads done before next iteration's DMA overwrites buf^1
  }

  // ---- epilogue (C layout: lane holds N=l16, rows M = r + 8*hf) ----
#pragma unroll
  for (int mt = 0; mt < 4; ++mt)
#pragma unroll
    for (int nt = 0; nt < 4; ++nt)
#pragma unroll
      for (int r = 0; r < 8; ++r) {
        int m = m0 + wm * 64 + mt * 16 + r + 8 * hf;
        int n = n0 + wn * 64 + nt * 16 + l16;
        float v = acc[mt][nt][r] + bias[n];
        if constexpr (MODE == 0) {
          ((float*)Cout)[(size_t)m * N + n] = v;
        } else if constexpr (MODE == 2) {
          v = fmaxf(v, 0.f);
          ((unsigned short*)Cout)[(size_t)m * N + n] = f32_to_bf16(v);
        } else if constexpr (MODE == 3) {  // Q/K: [b*NH+h][s][64]
          int h = n >> 6, d = n & 63, b = m >> 10, s = m & 1023;
          ((unsigned short*)Cout)[((size_t)(b * NHEAD + h) * SEQ + s) * HDIM + d] =
              f32_to_bf16(v);
        } else {                            // V transposed: [b*NH+h][d][s]
          int h = n >> 6, d = n & 63, b = m >> 10, s = m & 1023;
          ((unsigned short*)Cout)[((size_t)(b * NHEAD + h) * HDIM + d) * SEQ + s] =
              f32_to_bf16(v);
        }
      }
}

// ---------------------------------------------------------------------------
// Flash cross-attention. grid=(SEQ/64, B*NH), block=128 (4 waves).
// Each wave: 16 q-rows, streams KV in 32-token chunks, online softmax.
// ---------------------------------------------------------------------------
__global__ __launch_bounds__(128) void flash_attn(
    const unsigned short* __restrict__ Q,   // [96][1024][64] bf16
    const unsigned short* __restrict__ Kb,  // [96][1024][64] bf16
    const unsigned short* __restrict__ Vt,  // [96][64][1024] bf16
    float* __restrict__ ctx)                // [8][1024][768] f32
{
  __shared__ __align__(16) unsigned short lP[4][16 * LDSA];

  const int lane = threadIdx.x & 31;
  const int w    = threadIdx.x >> 5;
  const int l16  = lane & 15;
  const int hf   = lane >> 4;
  const int bh   = blockIdx.y;
  const int b    = bh / NHEAD;
  const int h    = bh - b * NHEAD;
  const int q0   = blockIdx.x * 64 + w * 16;

  const unsigned short* Qh = Q  + (size_t)bh * SEQ * HDIM;
  const unsigned short* Kh = Kb + (size_t)bh * SEQ * HDIM;
  const unsigned short* Vh = Vt + (size_t)bh * HDIM * SEQ;
  unsigned short* myP = lP[w];

  // Q A-operands, kept resident (2 chunks of K=32 over d)
  v16bf aQ[2];
#pragma unroll
  for (int c = 0; c < 2; ++c) {
    const unsigned short* p = Qh + (size_t)(q0 + l16) * HDIM + c * 32;
    v8bf lo = *(const v8bf*)(p + hf * 8);
    v8bf hi = *(const v8bf*)(p + 16 + hf * 8);
    aQ[c] = pack16(lo, hi);
  }

  v8f acc[4];
#pragma unroll
  for (int dt = 0; dt < 4; ++dt) acc[dt] = zero8();
  float mrow[8], lsum[8];
#pragma unroll
  for (int r = 0; r < 8; ++r) { mrow[r] = -1e30f; lsum[r] = 0.f; }

  for (int kv0 = 0; kv0 < SEQ; kv0 += 32) {
    // ---- scores: two 16x16 tiles ----
    v8f S[2] = {zero8(), zero8()};
#pragma unroll
    for (int nt = 0; nt < 2; ++nt) {
      const unsigned short* kp = Kh + (size_t)(kv0 + nt * 16 + l16) * HDIM;
#pragma unroll
      for (int c = 0; c < 2; ++c) {
        v8bf lo = *(const v8bf*)(kp + c * 32 + hf * 8);
        v8bf hi = *(const v8bf*)(kp + c * 32 + 16 + hf * 8);
        S[nt] = wmma_bf16(aQ[c], pack16(lo, hi), S[nt]);
      }
    }

    // ---- online softmax (row = r + 8*hf; cols distributed over 16 lanes) ----
#pragma unroll
    for (int r = 0; r < 8; ++r) {
      float s0 = S[0][r] * 0.125f;   // 1/sqrt(64)
      float s1 = S[1][r] * 0.125f;
      float mx = fmaxf(s0, s1);
      mx = fmaxf(mx, __shfl_xor(mx, 1, 32));
      mx = fmaxf(mx, __shfl_xor(mx, 2, 32));
      mx = fmaxf(mx, __shfl_xor(mx, 4, 32));
      mx = fmaxf(mx, __shfl_xor(mx, 8, 32));
      float mnew = fmaxf(mrow[r], mx);
      float sc = __expf(mrow[r] - mnew);
      mrow[r] = mnew;
      float p0 = __expf(s0 - mnew);
      float p1 = __expf(s1 - mnew);
      lsum[r] = lsum[r] * sc + p0 + p1;  // lane-partial; reduced at end
#pragma unroll
      for (int dt = 0; dt < 4; ++dt) acc[dt][r] *= sc;
      myP[(r + 8 * hf) * LDSA + l16]      = f32_to_bf16(p0);
      myP[(r + 8 * hf) * LDSA + 16 + l16] = f32_to_bf16(p1);
    }
    __syncthreads();

    // ---- P: C-layout -> A-layout through LDS ----
    v16bf aP;
    {
      const unsigned short* p = myP + l16 * LDSA;
      v8bf lo = *(const v8bf*)(p + hf * 8);
      v8bf hi = *(const v8bf*)(p + 16 + hf * 8);
      aP = pack16(lo, hi);
    }
    __syncthreads();

    // ---- ctx += P @ V  (B-operand = rows of V^T, contiguous over kv) ----
#pragma unroll
    for (int dt = 0; dt < 4; ++dt) {
      const unsigned short* vp =
          Vh + (size_t)(dt * 16 + l16) * SEQ + kv0 + hf * 8;
      v8bf lo = *(const v8bf*)(vp);
      v8bf hi = *(const v8bf*)(vp + 16);
      acc[dt] = wmma_bf16(aP, pack16(lo, hi), acc[dt]);
    }
  }

  // ---- finalize: reduce l over lanes, normalize, store f32 ctx ----
#pragma unroll
  for (int r = 0; r < 8; ++r) {
    float l = lsum[r];
    l += __shfl_xor(l, 1, 32);
    l += __shfl_xor(l, 2, 32);
    l += __shfl_xor(l, 4, 32);
    l += __shfl_xor(l, 8, 32);
    lsum[r] = 1.f / l;
  }
#pragma unroll
  for (int dt = 0; dt < 4; ++dt)
#pragma unroll
    for (int r = 0; r < 8; ++r) {
      int q = q0 + r + 8 * hf;
      int d = h * HDIM + dt * 16 + l16;
      ctx[((size_t)b * SEQ + q) * Hdim + d] = acc[dt][r] * lsum[r];
    }
}

// ---------------------------------------------------------------------------
// out = resid + LayerNorm(x)*g + beta ; optionally emit bf16 copy of out.
// one 768-row per 256-thread block.
// ---------------------------------------------------------------------------
__global__ __launch_bounds__(256) void ln_residual(
    const float* __restrict__ x, const float* __restrict__ resid,
    const float* __restrict__ g, const float* __restrict__ beta,
    float* __restrict__ out, unsigned short* __restrict__ obf) {
  __shared__ float rs[256], rq[256];
  const int row = blockIdx.x, t = threadIdx.x;
  const float* xr = x + (size_t)row * Hdim;
  float v0 = xr[t], v1 = xr[t + 256], v2 = xr[t + 512];
  rs[t] = v0 + v1 + v2;
  rq[t] = v0 * v0 + v1 * v1 + v2 * v2;
  __syncthreads();
  for (int off = 128; off > 0; off >>= 1) {
    if (t < off) { rs[t] += rs[t + off]; rq[t] += rq[t + off]; }
    __syncthreads();
  }
  float mean = rs[0] * (1.f / Hdim);
  float var  = rq[0] * (1.f / Hdim) - mean * mean;
  float rstd = rsqrtf(var + 1e-5f);
  const float* rr = resid + (size_t)row * Hdim;
  float v[3] = {v0, v1, v2};
#pragma unroll
  for (int i = 0; i < 3; ++i) {
    int c = t + i * 256;
    float y = (v[i] - mean) * rstd * g[c] + beta[c];
    float o = rr[c] + y;
    out[(size_t)row * Hdim + c] = o;
    if (obf) obf[(size_t)row * Hdim + c] = f32_to_bf16(o);
  }
}

// ---------------------------------------------------------------------------
extern "C" void kernel_launch(void* const* d_in, const int* in_sizes, int n_in,
                              void* d_out, int out_size, void* d_ws,
                              size_t ws_size, hipStream_t stream) {
  (void)in_sizes; (void)n_in; (void)out_size; (void)ws_size;
  const float* freq   = (const float*)d_in[0];
  const float* image  = (const float*)d_in[1];
  const float* Wq     = (const float*)d_in[2];
  const float* bq     = (const float*)d_in[3];
  const float* Wk     = (const float*)d_in[4];
  const float* bk     = (const float*)d_in[5];
  const float* Wv     = (const float*)d_in[6];
  const float* bv     = (const float*)d_in[7];
  const float* g_att  = (const float*)d_in[8];
  const float* b_att  = (const float*)d_in[9];
  const float* g_mlp  = (const float*)d_in[10];
  const float* b_mlp  = (const float*)d_in[11];
  const float* W1     = (const float*)d_in[12];
  const float* b1     = (const float*)d_in[13];
  const float* W2     = (const float*)d_in[14];
  const float* b2     = (const float*)d_in[15];
  float* out = (float*)d_out;
  char* ws = (char*)d_ws;

  constexpr size_t SZ_ACT_BF = (size_t)MROWS * Hdim * 2;   // 12.6 MB
  constexpr size_t SZ_W_BF   = (size_t)Hdim * Hdim * 2;
  constexpr size_t SZ_W1_BF  = (size_t)Hdim * FFDIM * 2;
  constexpr size_t SZ_ACT_F  = (size_t)MROWS * Hdim * 4;

  const size_t oXf = 0;
  const size_t oXi = oXf + SZ_ACT_BF;
  const size_t oQ  = oXi + SZ_ACT_BF;
  const size_t oK  = oQ  + SZ_ACT_BF;
  const size_t oV  = oK  + SZ_ACT_BF;
  const size_t oWq = oV  + SZ_ACT_BF;
  const size_t oWk = oWq + SZ_W_BF;
  const size_t oWv = oWk + SZ_W_BF;
  const size_t oW1 = oWv + SZ_W_BF;
  const size_t oW2 = oW1 + SZ_W1_BF;
  const size_t oCtx = oW2 + SZ_W1_BF;
  const size_t oH   = oCtx + SZ_ACT_F;
  const size_t oHb  = oH + SZ_ACT_F;
  const size_t oFF  = oXf;   // alias: Xf/Xi/Q/K dead before MLP1 writes (50.3 MB)
  const size_t oMlp = oCtx;  // alias: ctx dead before MLP2 writes

  unsigned short* Xf  = (unsigned short*)(ws + oXf);
  unsigned short* Xi  = (unsigned short*)(ws + oXi);
  unsigned short* Qb  = (unsigned short*)(ws + oQ);
  unsigned short* Kb  = (unsigned short*)(ws + oK);
  unsigned short* Vt  = (unsigned short*)(ws + oV);
  unsigned short* Wqt = (unsigned short*)(ws + oWq);
  unsigned short* Wkt = (unsigned short*)(ws + oWk);
  unsigned short* Wvt = (unsigned short*)(ws + oWv);
  unsigned short* W1t = (unsigned short*)(ws + oW1);
  unsigned short* W2t = (unsigned short*)(ws + oW2);
  float*          Ctx = (float*)(ws + oCtx);
  float*          Hbf = (float*)(ws + oH);
  unsigned short* Hb16= (unsigned short*)(ws + oHb);
  unsigned short* FFb = (unsigned short*)(ws + oFF);
  float*          Mlp = (float*)(ws + oMlp);

  // 1) converts
  cvt_bf16<<<4096, 256, 0, stream>>>(freq,  Xf, MROWS * Hdim);
  cvt_bf16<<<4096, 256, 0, stream>>>(image, Xi, MROWS * Hdim);
  tcvt_bf16<<<2048, 256, 0, stream>>>(Wq, Wqt, Hdim, Hdim);
  tcvt_bf16<<<2048, 256, 0, stream>>>(Wk, Wkt, Hdim, Hdim);
  tcvt_bf16<<<2048, 256, 0, stream>>>(Wv, Wvt, Hdim, Hdim);
  tcvt_bf16<<<4096, 256, 0, stream>>>(W1, W1t, Hdim, FFDIM);
  tcvt_bf16<<<4096, 256, 0, stream>>>(W2, W2t, FFDIM, Hdim);

  // 2) QKV projections (WMMA, TDM-fed)
  dim3 gP(Hdim / 128, MROWS / 256);
  gemm_bf16<3><<<gP, 256, 0, stream>>>(Xf, Wqt, bq, Qb, MROWS, Hdim, Hdim);
  gemm_bf16<3><<<gP, 256, 0, stream>>>(Xi, Wkt, bk, Kb, MROWS, Hdim, Hdim);
  gemm_bf16<4><<<gP, 256, 0, stream>>>(Xi, Wvt, bv, Vt, MROWS, Hdim, Hdim);

  // 3) flash attention (WMMA)
  flash_attn<<<dim3(SEQ / 64, BATCH * NHEAD), 128, 0, stream>>>(Qb, Kb, Vt, Ctx);

  // 4) h = freq + LN(ctx)  (+ bf16 copy for MLP)
  ln_residual<<<MROWS, 256, 0, stream>>>(Ctx, freq, g_att, b_att, Hbf, Hb16);

  // 5) MLP (WMMA, TDM-fed)
  gemm_bf16<2><<<dim3(FFDIM / 128, MROWS / 256), 256, 0, stream>>>(
      Hb16, W1t, b1, FFb, MROWS, FFDIM, Hdim);
  gemm_bf16<0><<<dim3(Hdim / 128, MROWS / 256), 256, 0, stream>>>(
      FFb, W2t, b2, Mlp, MROWS, Hdim, FFDIM);

  // 6) out = h + LN(mlp)
  ln_residual<<<MROWS, 256, 0, stream>>>(Mlp, Hbf, g_mlp, b_mlp, out, nullptr);
}
